// LoRAQKVWrapper_26250840113720
// MI455X (gfx1250) — matrix operand order repaired and checked by reading, not compiled, not run on hardware
//
#include <hip/hip_runtime.h>

typedef __attribute__((ext_vector_type(16))) _Float16 v16h;
typedef __attribute__((ext_vector_type(8)))  _Float16 v8h;
typedef __attribute__((ext_vector_type(8)))  float    v8f;

#define M_TOT 8192
#define N_TOT 4096
#define K_TOT 4096
#define R_LORA 16
#define BM 128
#define BN 128
#define BK 32
#define LDT 40                      // f16 elems per LDS row: 80B, 16B aligned
#define NK (K_TOT / BK)             // 128
#define BUFE ((BM + BN) * LDT)      // f16 elems per LDS buffer (10240)
#define BUFB (BUFE * 2)             // bytes per LDS buffer (20480)
#define NBUF 3                      // triple-buffered pipeline

// ---------------------------------------------------------------------------
// async global -> LDS copy (CDNA5), 16 bytes per lane, ASYNCcnt-tracked
// ---------------------------------------------------------------------------
__device__ __forceinline__ void async_b128(unsigned lds_addr, const _Float16* g) {
  asm volatile("global_load_async_to_lds_b128 %0, %1, off"
               :: "v"(lds_addr), "v"(g)
               : "memory");
}

// ---------------------------------------------------------------------------
// Kernel 1: convert x rows fp32 -> f16 AND compute LoRA stage 1:
//   xa[p][m][r] = sum_d x[m,d] * A_p[r,d]      (fp32, one wave per row m)
// ---------------------------------------------------------------------------
__global__ __launch_bounds__(256) void conv_x_lora(
    const float* __restrict__ x,
    const float* __restrict__ qA, const float* __restrict__ kA,
    const float* __restrict__ vA,
    _Float16* __restrict__ xh, float* __restrict__ xa)
{
  int wave = (int)((blockIdx.x * blockDim.x + threadIdx.x) >> 5);
  int lane = (int)(threadIdx.x & 31);
  if (wave >= M_TOT) return;
  const float* xr = x + (size_t)wave * K_TOT;
  _Float16* xo = xh + (size_t)wave * K_TOT;
  float acc[48];
#pragma unroll
  for (int t = 0; t < 48; ++t) acc[t] = 0.f;
  for (int d = lane; d < K_TOT; d += 32) {
    float xv = xr[d];
    xo[d] = (_Float16)xv;
#pragma unroll
    for (int r = 0; r < 16; ++r) acc[r]      += xv * qA[r * K_TOT + d];
#pragma unroll
    for (int r = 0; r < 16; ++r) acc[16 + r] += xv * kA[r * K_TOT + d];
#pragma unroll
    for (int r = 0; r < 16; ++r) acc[32 + r] += xv * vA[r * K_TOT + d];
  }
#pragma unroll
  for (int t = 0; t < 48; ++t) {
    float v = acc[t];
    for (int off = 16; off > 0; off >>= 1) v += __shfl_xor(v, off, 32);
    if (lane == 0) {
      int p = t >> 4, r = t & 15;
      xa[((size_t)p * M_TOT + wave) * R_LORA + r] = v;
    }
  }
}

// ---------------------------------------------------------------------------
// Kernel 2: convert the three weight matrices fp32 -> f16 (packed [3,N,K])
// ---------------------------------------------------------------------------
__global__ __launch_bounds__(256) void conv_w(
    const float* __restrict__ qW, const float* __restrict__ kW,
    const float* __restrict__ vW, _Float16* __restrict__ wh)
{
  const float* src = (blockIdx.z == 0) ? qW : ((blockIdx.z == 1) ? kW : vW);
  _Float16* dst = wh + (size_t)blockIdx.z * N_TOT * K_TOT;
  size_t i = ((size_t)blockIdx.x * blockDim.x + threadIdx.x) * 8;
  float4 a = *(const float4*)(src + i);
  float4 b = *(const float4*)(src + i + 4);
  v8h h;
  h[0] = (_Float16)a.x; h[1] = (_Float16)a.y; h[2] = (_Float16)a.z; h[3] = (_Float16)a.w;
  h[4] = (_Float16)b.x; h[5] = (_Float16)b.y; h[6] = (_Float16)b.z; h[7] = (_Float16)b.w;
  *(v8h*)(dst + i) = h;
}

// ---------------------------------------------------------------------------
// Kernel 3: fused GEMM  out[p] = Xh @ Wh[p]^T + xa[p] @ (Bw[p])^T
//   128x128 tile / workgroup, 8 waves, each wave = 4x2 WMMA tiles (64x32).
//   Triple-buffered LDS fed by global_load_async_to_lds_b128; in-order
//   ASYNCcnt lets "s_wait_asynccnt 4" retire tile kb while kb+1 is in flight.
// ---------------------------------------------------------------------------
__global__ __launch_bounds__(256) void lora_qkv_gemm(
    const _Float16* __restrict__ xh,   // [M,K]
    const _Float16* __restrict__ wh,   // [3,N,K]
    const float* __restrict__ xa,      // [3,M,R]
    const float* __restrict__ qB, const float* __restrict__ kB,
    const float* __restrict__ vB,
    float* __restrict__ out)           // [3,M,N]
{
  __shared__ _Float16 lds[NBUF * BUFE];   // 60 KiB

  const int p  = (int)blockIdx.z;
  const int n0 = (int)blockIdx.x * BN;
  const int m0 = (int)blockIdx.y * BM;
  const _Float16* wp = wh + (size_t)p * N_TOT * K_TOT;

  const int tid     = (int)threadIdx.x;
  const int lane    = tid & 31;
  const int waveId  = tid >> 5;
  const int waveRow = waveId & 1;   // 2 row groups of 64 rows
  const int waveCol = waveId >> 1;  // 4 col groups of 32 cols
  const int hl      = lane >> 4;    // half of wave
  const int lr      = lane & 15;

  // Async copy plan: 4 lanes per tile row, 16B per lane, 4 passes of 64 rows.
  const int chunk = tid & 3;        // 16B chunk within row
  const int rbase = tid >> 2;       // row 0..63, +64 per pass
  const _Float16* gsrc[4];
  unsigned ldst[4];
#pragma unroll
  for (int pass = 0; pass < 4; ++pass) {
    int row = rbase + pass * 64;
    const _Float16* base = (row < BM)
        ? (xh + (size_t)(m0 + row) * K_TOT)
        : (wp + (size_t)(n0 + (row - BM)) * K_TOT);
    gsrc[pass] = base + chunk * 8;
    ldst[pass] = (unsigned)(size_t)(const void*)&lds[row * LDT + chunk * 8];
  }

  v8f acc[4][2];
#pragma unroll
  for (int i = 0; i < 4; ++i)
#pragma unroll
    for (int j = 0; j < 2; ++j)
      acc[i][j] = (v8f){0.f, 0.f, 0.f, 0.f, 0.f, 0.f, 0.f, 0.f};

  // prologue: start filling buffers 0 (kb=0) and 1 (kb=1)
#pragma unroll
  for (int pass = 0; pass < 4; ++pass)
    async_b128(ldst[pass], gsrc[pass]);
#pragma unroll
  for (int pass = 0; pass < 4; ++pass)
    async_b128(ldst[pass] + BUFB, gsrc[pass] + BK);

  unsigned issueB = 2 * BUFB;   // byte offset of LDS buffer receiving kb+2
  unsigned curE   = 0;          // element offset of LDS buffer holding kb

  for (int kb = 0; kb < NK; ++kb) {
    // retire batch kb (in-order completion); batch kb+1 may stay in flight
    if (kb + 1 < NK) asm volatile("s_wait_asynccnt 0x4" ::: "memory");
    else             asm volatile("s_wait_asynccnt 0x0" ::: "memory");
    // publish batch kb to all waves; also fences last readers of the buffer
    // that batch kb+2 will overwrite (syncthreads lowers with s_wait_dscnt 0)
    __syncthreads();

    if (kb + 2 < NK) {
      const size_t gofs = (size_t)(kb + 2) * BK;
#pragma unroll
      for (int pass = 0; pass < 4; ++pass)
        async_b128(ldst[pass] + issueB, gsrc[pass] + gofs);
    }

    const _Float16* cur = lds + curE;

    // A fragments: lane lr holds row (base+lr); hl=0 -> K 0-7 & 16-23,
    //              hl=1 -> K 8-15 & 24-31   (ISA 16-bit A 16x32 layout)
    v16h afr[4];
#pragma unroll
    for (int i = 0; i < 4; ++i) {
      const _Float16* src = cur + (waveRow * 64 + i * 16 + lr) * LDT;
      union { v16h v; uint4 u[2]; } t;
      t.u[0] = *(const uint4*)(src + hl * 8);
      t.u[1] = *(const uint4*)(src + 16 + hl * 8);
      afr[i] = t.v;
    }
    // B fragments: lane lr holds col (=W row) base+lr; hl=0 -> K 0-7 & 16-23,
    //              hl=1 -> K 8-15 & 24-31  (same K order as A)
    v16h bfr[2];
#pragma unroll
    for (int j = 0; j < 2; ++j) {
      const _Float16* src = cur + (BM + waveCol * 32 + j * 16 + lr) * LDT;
      union { v16h v; uint4 u[2]; } t;
      t.u[0] = *(const uint4*)(src + hl * 8);
      t.u[1] = *(const uint4*)(src + 16 + hl * 8);
      bfr[j] = t.v;
    }

#pragma unroll
    for (int i = 0; i < 4; ++i)
#pragma unroll
      for (int j = 0; j < 2; ++j)
        acc[i][j] = __builtin_amdgcn_wmma_f32_16x16x32_f16(
            false, afr[i], false, bfr[j], (short)0, acc[i][j], false, false);

    issueB += BUFB; if (issueB == NBUF * BUFB) issueB = 0;
    curE   += BUFE; if (curE   == NBUF * BUFE) curE   = 0;
  }

  // Epilogue: add rank-16 LoRA term and store (C layout: VGPR v -> row v+8*hl)
  const float* bwp = (p == 0) ? qB : ((p == 1) ? kB : vB);
  float* outp = out + (size_t)p * M_TOT * N_TOT;
  const float* xap = xa + (size_t)p * M_TOT * R_LORA;

#pragma unroll
  for (int j = 0; j < 2; ++j) {
    int n = n0 + waveCol * 32 + j * 16 + lr;
    const float4* bw4 = (const float4*)(bwp + (size_t)n * R_LORA);
    float4 bw0 = bw4[0], bw1 = bw4[1], bw2 = bw4[2], bw3 = bw4[3];
#pragma unroll
    for (int i = 0; i < 4; ++i) {
      int mbase = m0 + waveRow * 64 + i * 16 + hl * 8;
#pragma unroll
      for (int v = 0; v < 8; ++v) {
        int m = mbase + v;
        const float4* xr4 = (const float4*)(xap + (size_t)m * R_LORA);
        float4 x0 = xr4[0], x1 = xr4[1], x2 = xr4[2], x3 = xr4[3];
        float l = x0.x * bw0.x + x0.y * bw0.y + x0.z * bw0.z + x0.w * bw0.w
                + x1.x * bw1.x + x1.y * bw1.y + x1.z * bw1.z + x1.w * bw1.w
                + x2.x * bw2.x + x2.y * bw2.y + x2.z * bw2.z + x2.w * bw2.w
                + x3.x * bw3.x + x3.y * bw3.y + x3.z * bw3.z + x3.w * bw3.w;
        outp[(size_t)m * N_TOT + n] = acc[i][j][v] + l;
      }
    }
  }
}

// ---------------------------------------------------------------------------
extern "C" void kernel_launch(void* const* d_in, const int* in_sizes, int n_in,
                              void* d_out, int out_size, void* d_ws, size_t ws_size,
                              hipStream_t stream) {
  (void)in_sizes; (void)n_in; (void)out_size; (void)ws_size;
  const float* x  = (const float*)d_in[0];
  const float* qW = (const float*)d_in[1];
  const float* kW = (const float*)d_in[2];
  const float* vW = (const float*)d_in[3];
  const float* qA = (const float*)d_in[4];
  const float* qB = (const float*)d_in[5];
  const float* kA = (const float*)d_in[6];
  const float* kB = (const float*)d_in[7];
  const float* vA = (const float*)d_in[8];
  const float* vB = (const float*)d_in[9];
  float* out = (float*)d_out;

  char* ws = (char*)d_ws;
  _Float16* xh = (_Float16*)ws;                                   // 64 MiB
  _Float16* wh = (_Float16*)(ws + (size_t)M_TOT * K_TOT * 2);     // 96 MiB
  float*    xa = (float*)(ws + (size_t)M_TOT * K_TOT * 2
                             + (size_t)3 * N_TOT * K_TOT * 2);    // 1.5 MiB

  conv_w<<<dim3((N_TOT * (size_t)K_TOT) / (256 * 8), 1, 3), 256, 0, stream>>>(
      qW, kW, vW, wh);
  conv_x_lora<<<dim3(M_TOT / 8), 256, 0, stream>>>(x, qA, kA, vA, xh, xa);
  lora_qkv_gemm<<<dim3(N_TOT / BN, M_TOT / BM, 3), 256, 0, stream>>>(
      xh, wh, xa, qB, kB, vB, out);
}
